// CrossMamba_44736379355723
// MI455X (gfx1250) — compile-verified, matrix operand order checked by reading, not run on hardware
//
#include <hip/hip_runtime.h>
#include <hip/hip_bf16.h>
#include <math.h>

typedef __bf16 bf16_t;
typedef __attribute__((ext_vector_type(8)))  __bf16 v8bf;
typedef __attribute__((ext_vector_type(16))) __bf16 v16bf;
typedef __attribute__((ext_vector_type(8)))  float  v8f;

#define B_SZ    2
#define L_CTX   2048
#define L_Q     2048
#define L_TOT   4096
#define DMODEL  512
#define DINNER  1024
#define NXZ     2048   // 2*DINNER
#define DTRANK  32
#define DSTATE  16
#define NDBL    64     // DTRANK + 2*DSTATE

// LDS panel geometry: rows padded to 80B to spread across the 64 banks.
#define ROWB      80u
#define A_BYTES   (128u * ROWB)          // 10240
#define B_BYTES   (64u * ROWB)           // 5120
#define BUF_BYTES (A_BYTES + B_BYTES)    // 15360

// ---- CDNA5 async global->LDS (ASYNCcnt tracked) -------------------------
__device__ __forceinline__ void async_ld_b128(unsigned lds_addr, const void* gptr) {
    asm volatile("global_load_async_to_lds_b128 %0, %1, off"
                 :: "v"(lds_addr), "v"(gptr) : "memory");
}
#define WAIT_ASYNC_3() asm volatile("s_wait_asynccnt 0x3" ::: "memory")
#define WAIT_ASYNC_0() asm volatile("s_wait_asynccnt 0x0" ::: "memory")

// ---------------------------------------------------------------- utilities
__global__ void k_cvt_bf16(const float* __restrict__ src, bf16_t* __restrict__ dst, int n) {
    int i = blockIdx.x * blockDim.x + threadIdx.x;
    if (i < n) dst[i] = (bf16_t)src[i];
}

// x = concat(context+seg_c, query+seg_q)  ->  bf16, (B, L_TOT, DMODEL)
__global__ void k_build_x(const float* __restrict__ query, const float* __restrict__ context,
                          const float* __restrict__ seg_c, const float* __restrict__ seg_q,
                          bf16_t* __restrict__ x_bf) {
    size_t i = (size_t)blockIdx.x * blockDim.x + threadIdx.x;
    if (i >= (size_t)B_SZ * L_TOT * DMODEL) return;
    int k = (int)(i % DMODEL);
    size_t bl = i / DMODEL;
    int l = (int)(bl % L_TOT);
    int b = (int)(bl / L_TOT);
    float v;
    if (l < L_CTX) v = context[((size_t)b * L_CTX + l) * DMODEL + k] + seg_c[k];
    else           v = query[((size_t)b * L_Q + (l - L_CTX)) * DMODEL + k] + seg_q[k];
    x_bf[i] = (bf16_t)v;
}

// ------------------------------------------------------- tiled WMMA GEMM
// C[m,n] = sum_k A[m,k] * Bw[n,k]  (A @ W.T, weights row-major N x K)
// Block tile 128(M) x 64(N); 8 waves, each wave a 32x32 tile (2x2 WMMA accs).
// A/B panels double-buffered in LDS via global_load_async_to_lds_b128.
// Per-thread staging addresses are hoisted out of the K-loop: the issue step
// is 3 async b128 ops + 3 pointer bumps, no divergence, no multiplies.
// epi==0: store f32 C (+ optional bf16 copy Cb)
// epi==1: scatter 0.5*acc into Out for query rows (dir0 '=', dir1 '+=').
__global__ void __launch_bounds__(256)
k_gemm_bf16(const bf16_t* __restrict__ A, int lda,
            const bf16_t* __restrict__ Bw, int ldb,
            int M, int N, int K,
            float* __restrict__ C, bf16_t* __restrict__ Cb,
            int epi, float* __restrict__ Out, int dir) {
    __shared__ __attribute__((aligned(16))) unsigned char smem[2 * BUF_BYTES];

    const int tilesN = N >> 6;                       // 64-wide N tiles
    const int btm = (blockIdx.x / tilesN) << 7;      // 128-row block tile
    const int btn = (blockIdx.x % tilesN) << 6;

    const int tid  = threadIdx.x;
    const int lane = tid & 31;
    const int w    = tid >> 5;                       // 0..7
    const int wm   = w >> 1;                         // 0..3 (32-row strip)
    const int wn   = w & 1;                          // 0..1 (32-col strip)
    const int row  = lane & 15;
    const unsigned kh2 = (unsigned)((lane >> 4) * 16);   // byte offset of K-half

    const unsigned sbase = (unsigned)(unsigned long long)(void*)&smem[0];

    // ---- staging descriptors, computed ONCE (3 b128 lanes / thread) ------
    // 768 16B chunks per k-chunk: rows 0..127 -> A panel, 128..191 -> B panel.
    const unsigned char* gp[3];
    unsigned             lo[3];
#pragma unroll
    for (int i = 0; i < 3; ++i) {
        const int c   = i * 256 + tid;
        const int r16 = c >> 2;
        const int col = c & 3;
        if (r16 < 128) {
            lo[i] = (unsigned)r16 * ROWB + (unsigned)col * 16u;
            gp[i] = (const unsigned char*)(A + (size_t)(btm + r16) * lda + col * 8);
        } else {
            const int rb = r16 - 128;
            lo[i] = A_BYTES + (unsigned)rb * ROWB + (unsigned)col * 16u;
            gp[i] = (const unsigned char*)(Bw + (size_t)(btn + rb) * ldb + col * 8);
        }
    }

    auto issue_chunk = [&](int buf) {
        const unsigned bo = sbase + (unsigned)buf * BUF_BYTES;
#pragma unroll
        for (int i = 0; i < 3; ++i) {
            async_ld_b128(bo + lo[i], gp[i]);
            gp[i] += 64;                              // next 32-wide K chunk
        }
    };

    auto ldtile = [&](unsigned rowbase) -> v16bf {
        v8bf lo8 = *(const v8bf*)&smem[rowbase + kh2];
        v8bf hi8 = *(const v8bf*)&smem[rowbase + 32u + kh2];
        return __builtin_shufflevector(lo8, hi8, 0,1,2,3,4,5,6,7,8,9,10,11,12,13,14,15);
    };

    v8f acc[2][2] = {};
    const int NK = K >> 5;

    issue_chunk(0);
    for (int kc = 0; kc < NK; ++kc) {
        const int buf = kc & 1;
        const bool hasNext = (kc + 1) < NK;
        if (hasNext) { issue_chunk(buf ^ 1); WAIT_ASYNC_3(); }
        else         { WAIT_ASYNC_0(); }
        __syncthreads();

        const unsigned base = (unsigned)buf * BUF_BYTES;
        const unsigned ab = base + (unsigned)(wm * 32 + row) * ROWB;
        const unsigned bb = base + A_BYTES + (unsigned)(wn * 32 + row) * ROWB;
        v16bf a0 = ldtile(ab);
        v16bf a1 = ldtile(ab + 16u * ROWB);
        v16bf b0 = ldtile(bb);
        v16bf b1 = ldtile(bb + 16u * ROWB);
        acc[0][0] = __builtin_amdgcn_wmma_f32_16x16x32_bf16(false, a0, false, b0, (short)0, acc[0][0], false, false);
        acc[0][1] = __builtin_amdgcn_wmma_f32_16x16x32_bf16(false, a0, false, b1, (short)0, acc[0][1], false, false);
        acc[1][0] = __builtin_amdgcn_wmma_f32_16x16x32_bf16(false, a1, false, b0, (short)0, acc[1][0], false, false);
        acc[1][1] = __builtin_amdgcn_wmma_f32_16x16x32_bf16(false, a1, false, b1, (short)0, acc[1][1], false, false);
        __syncthreads();
    }

    // D layout: element r -> M = r + 8*(lane/16), N = lane%16
    const int mofs = (lane >> 4) << 3;
#pragma unroll
    for (int ti = 0; ti < 2; ++ti) {
#pragma unroll
        for (int tj = 0; tj < 2; ++tj) {
            const int m0 = btm + wm * 32 + ti * 16 + mofs;
            const int n0 = btn + wn * 32 + tj * 16 + (lane & 15);
            if (epi == 0) {
#pragma unroll
                for (int r = 0; r < 8; ++r) {
                    size_t o = (size_t)(m0 + r) * N + n0;
                    C[o] = acc[ti][tj][r];
                    if (Cb) Cb[o] = (bf16_t)acc[ti][tj][r];
                }
            } else {
#pragma unroll
                for (int r = 0; r < 8; ++r) {
                    int m   = m0 + r;
                    int bb2 = m / L_TOT;
                    int tl  = m - bb2 * L_TOT;
                    int tp  = dir ? (L_TOT - 1 - tl) : tl;  // logical -> physical
                    if (tp >= L_CTX) {
                        size_t o = ((size_t)bb2 * L_Q + (tp - L_CTX)) * N + n0;
                        float v = 0.5f * acc[ti][tj][r];
                        if (dir == 0) Out[o] = v; else Out[o] += v;
                    }
                }
            }
        }
    }
}

// ------------------------------------------- depthwise causal conv + SiLU
__global__ void k_conv_silu(const float* __restrict__ xz,
                            const float* __restrict__ conv_w,
                            const float* __restrict__ conv_b,
                            bf16_t* __restrict__ xc_bf, int dir) {
    size_t i = (size_t)blockIdx.x * blockDim.x + threadIdx.x;
    if (i >= (size_t)B_SZ * L_TOT * DINNER) return;
    int d = (int)(i % DINNER);
    size_t bl = i / DINNER;
    int tl = (int)(bl % L_TOT);
    int b  = (int)(bl / L_TOT);
    float acc = conv_b[d];
#pragma unroll
    for (int j = 0; j < 4; ++j) {
        int ts = tl - 3 + j;
        if (ts >= 0) {
            int tp = dir ? (L_TOT - 1 - ts) : ts;
            acc += conv_w[d * 4 + j] * xz[((size_t)b * L_TOT + tp) * NXZ + d];
        }
    }
    float s = acc / (1.0f + __expf(-acc));   // SiLU
    xc_bf[((size_t)b * L_TOT + tl) * DINNER + d] = (bf16_t)s;
}

// ----------------------------------------------------- selective scan (SSM)
__global__ void k_scan(const float* __restrict__ dtpre, const float* __restrict__ b_dt,
                       const float* __restrict__ A_log, const float* __restrict__ Dp,
                       const float* __restrict__ dbl,   const bf16_t* __restrict__ xc_bf,
                       const float* __restrict__ xz,    bf16_t* __restrict__ y_bf, int dir) {
    __shared__ float sBC[32];
    const int tid = threadIdx.x;                 // 64 threads/block
    const int b   = blockIdx.x >> 4;             // 16 blocks per batch entry
    const int d   = ((blockIdx.x & 15) << 6) + tid;

    float Aneg[DSTATE];
#pragma unroll
    for (int s = 0; s < DSTATE; ++s) Aneg[s] = -__expf(A_log[d * DSTATE + s]);
    const float Dv  = Dp[d];
    const float bdt = b_dt[d];
    float h[DSTATE];
#pragma unroll
    for (int s = 0; s < DSTATE; ++s) h[s] = 0.f;

    for (int t = 0; t < L_TOT; ++t) {
        const size_t rbase = (size_t)b * L_TOT + t;
        if (tid < 32) sBC[tid] = dbl[rbase * NDBL + DTRANK + tid];  // B_t | C_t
        __syncthreads();

        float dtv = dtpre[rbase * DINNER + d] + bdt;
        dtv = (dtv > 20.f) ? dtv : log1pf(__expf(dtv));             // softplus
        const float xv = (float)xc_bf[rbase * DINNER + d];
        const float dx = dtv * xv;
        float y = 0.f;
#pragma unroll
        for (int s = 0; s < DSTATE; ++s) {
            const float dA = __expf(dtv * Aneg[s]);
            h[s] = dA * h[s] + dx * sBC[s];
            y += h[s] * sBC[16 + s];
        }
        y += xv * Dv;
        const int tp = dir ? (L_TOT - 1 - t) : t;
        const float zv = xz[((size_t)b * L_TOT + tp) * NXZ + DINNER + d];
        y *= zv / (1.f + __expf(-zv));                              // * silu(z)
        y_bf[rbase * DINNER + d] = (bf16_t)y;
        __syncthreads();
    }
}

// ------------------------------------------------------------------ driver
extern "C" void kernel_launch(void* const* d_in, const int* in_sizes, int n_in,
                              void* d_out, int out_size, void* d_ws, size_t ws_size,
                              hipStream_t stream) {
    const float* query   = (const float*)d_in[0];
    const float* context = (const float*)d_in[1];
    const float* seg_c   = (const float*)d_in[2];
    const float* seg_q   = (const float*)d_in[3];
    const float* W_in    = (const float*)d_in[4];
    const float* conv_w  = (const float*)d_in[5];
    const float* conv_b  = (const float*)d_in[6];
    const float* W_x     = (const float*)d_in[7];
    const float* W_dt    = (const float*)d_in[8];
    const float* b_dt    = (const float*)d_in[9];
    const float* A_log   = (const float*)d_in[10];
    const float* D_in    = (const float*)d_in[11];
    const float* W_out   = (const float*)d_in[12];
    float* out = (float*)d_out;

    char* p = (char*)d_ws;
    auto alloc = [&](size_t bytes) { char* r = p; p += (bytes + 255) & ~(size_t)255; return r; };
    const size_t M = (size_t)B_SZ * L_TOT;                         // 8192 rows

    float*  xz      = (float*) alloc(M * NXZ * 4);                 // 64 MB
    bf16_t* x_bf    = (bf16_t*)alloc(M * DMODEL * 2);              //  8 MB
    bf16_t* Win_bf  = (bf16_t*)alloc((size_t)NXZ * DMODEL * 2);
    bf16_t* xc_bf   = (bf16_t*)alloc(M * DINNER * 2);              // 16 MB
    bf16_t* Wx_bf   = (bf16_t*)alloc((size_t)NDBL * DINNER * 2);
    float*  dbl     = (float*) alloc(M * NDBL * 4);
    bf16_t* dbl_bf  = (bf16_t*)alloc(M * NDBL * 2);
    bf16_t* Wdt_bf  = (bf16_t*)alloc((size_t)DINNER * DTRANK * 2);
    float*  dtpre   = (float*) alloc(M * DINNER * 4);              // 32 MB
    bf16_t* y_bf    = (bf16_t*)alloc(M * DINNER * 2);              // 16 MB
    bf16_t* Wout_bf = (bf16_t*)alloc((size_t)DMODEL * DINNER * 2);

    auto cvt = [&](const float* s, bf16_t* dst, int n) {
        k_cvt_bf16<<<(n + 255) / 256, 256, 0, stream>>>(s, dst, n);
    };
    cvt(W_in,  Win_bf,  NXZ * DMODEL);
    cvt(W_x,   Wx_bf,   NDBL * DINNER);
    cvt(W_dt,  Wdt_bf,  DINNER * DTRANK);
    cvt(W_out, Wout_bf, DMODEL * DINNER);

    {
        int n = (int)(M * DMODEL);
        k_build_x<<<(n + 255) / 256, 256, 0, stream>>>(query, context, seg_c, seg_q, x_bf);
    }

    auto gemm = [&](const bf16_t* A, int lda, const bf16_t* Bw, int ldb,
                    int Mm, int Nn, int Kk, float* C, bf16_t* Cb, int epi, int dir) {
        int blocks = (Mm >> 7) * (Nn >> 6);                        // 128x64 block tiles
        k_gemm_bf16<<<blocks, 256, 0, stream>>>(A, lda, Bw, ldb, Mm, Nn, Kk,
                                                C, Cb, epi, out, dir);
    };

    // xz = x @ W_in.T   (direction independent)
    gemm(x_bf, DMODEL, Win_bf, DMODEL, (int)M, NXZ, DMODEL, xz, nullptr, 0, 0);

    for (int dir = 0; dir < 2; ++dir) {
        {
            size_t n = M * DINNER;
            k_conv_silu<<<(int)((n + 255) / 256), 256, 0, stream>>>(xz, conv_w, conv_b,
                                                                    xc_bf, dir);
        }
        // dbl = xc @ W_x.T  (f32 for B/C, bf16 copy feeds the dt GEMM)
        gemm(xc_bf, DINNER, Wx_bf, DINNER, (int)M, NDBL, DINNER, dbl, dbl_bf, 0, 0);
        // dtpre = dbl[:, :DTRANK] @ W_dt.T  (lda=64 selects the first 32 cols)
        gemm(dbl_bf, NDBL, Wdt_bf, DTRANK, (int)M, DINNER, DTRANK, dtpre, nullptr, 0, 0);
        // sequential SSM scan + gating -> y bf16
        k_scan<<<B_SZ * 16, 64, 0, stream>>>(dtpre, b_dt, A_log, D_in, dbl,
                                             xc_bf, xz, y_bf, dir);
        // out GEMM with fused 0.5*(fwd+bwd) scatter into d_out
        gemm(y_bf, DINNER, Wout_bf, DINNER, (int)M, DMODEL, DINNER, nullptr, nullptr, 1, dir);
    }
}
